// KAN_GNN_80693845557834
// MI455X (gfx1250) — matrix operand order repaired and checked by reading, not compiled, not run on hardware
//
#include <hip/hip_runtime.h>
#include <hip/hip_bf16.h>

#define NN     10000
#define EREAL  160000
#define ETOT   170000

typedef __attribute__((ext_vector_type(16))) __bf16       v16bf;
typedef __attribute__((ext_vector_type(8)))  float        v8f;
typedef __attribute__((ext_vector_type(4)))  unsigned int v4u;

union Frag16 { v16bf v; v4u q[2]; unsigned short u[16]; };

__device__ __forceinline__ unsigned short f2bf(float f) {
  unsigned int u = __float_as_uint(f);
  u += 0x7FFFu + ((u >> 16) & 1u);           // round-to-nearest-even
  return (unsigned short)(u >> 16);
}

// wave-scope LDS producer->consumer sync (per-wave LDS tiles, no block
// barrier needed). CDNA5 split wait counter: DScnt covers all LDS traffic.
__device__ __forceinline__ void wave_lds_sync() {
  __builtin_amdgcn_wave_barrier();
  asm volatile("s_wait_dscnt 0" ::: "memory");
  __builtin_amdgcn_wave_barrier();
}

__device__ __forceinline__ void atomic_add_f32(float* p, float v) {
  __hip_atomic_fetch_add(p, v, __ATOMIC_RELAXED, __HIP_MEMORY_SCOPE_AGENT);
}

__device__ __forceinline__ void load_edge(const int* __restrict__ ei, int e, int& r, int& c) {
  if (e < EREAL) { r = ei[e]; c = ei[EREAL + e]; }
  else           { r = e - EREAL; c = r; }      // appended self loops
}

// Uniform-knot cubic B-spline: at x in [t_{j0}, t_{j0+1}) only bases
// j0-3..j0 are nonzero, with the cardinal cubic polynomials of
// u = (x-t0)/h - j0.  Zero the 8-slot group (one 16B store), then write
// the <=4 nonzero bf16 values at computed LDS offsets.
__device__ __forceinline__ void basis_store(unsigned short* __restrict__ dst8,
                                            float x, float t0, float inv_h) {
  v4u z = {0u, 0u, 0u, 0u};
  *(v4u*)dst8 = z;
  float s  = (x - t0) * inv_h;
  float jf = floorf(s);
  int j0 = (int)jf;
  j0 = j0 < 0 ? 0 : (j0 > 10 ? 10 : j0);
  float u  = s - (float)j0;
  float um = 1.f - u;
  float u2 = u * u, u3 = u2 * u;
  float um3 = um * um * um;
  const float c6 = 1.f / 6.f;
  float B0 = um3 * c6;
  float B1 = (3.f * u3 - 6.f * u2 + 4.f) * c6;
  float B2 = (-3.f * u3 + 3.f * u2 + 3.f * u + 1.f) * c6;
  float B3 = u3 * c6;
  int base = j0 - 3;                    // in [-3, 7]
  if (base >= 0)                 dst8[base]     = f2bf(B0);
  if (base + 1 >= 0 && base < 7) dst8[base + 1] = f2bf(B1);
  if (base + 2 >= 0 && base < 6) dst8[base + 2] = f2bf(B2);
  if (j0 < 8)                    dst8[j0]       = f2bf(B3);
}

// A fragment, contiguous K order matching B: lanes0-15 K={0..15}(+kk),
// lanes16-31 K={16..31}(+kk). Two contiguous 16B LDS reads.
__device__ __forceinline__ void load_afrag(Frag16& a, const unsigned short* asw,
                                           int mA, int hi, int kk) {
  const unsigned short* ap = asw + mA * 128 + kk + hi * 16;
  a.q[0] = *(const v4u*)ap;
  a.q[1] = *(const v4u*)(ap + 8);
}

// B pre-swizzled by prep kernel so each lane reads its 16 bf16 contiguously.
__device__ __forceinline__ void load_bfrag(Frag16& b, const unsigned short* __restrict__ Bmat,
                                           int kb, int NT, int nt, int lane) {
  const v4u* bp = (const v4u*)(Bmat + ((((size_t)kb * NT + nt) * 32 + lane) << 4));
  b.q[0] = bp[0];
  b.q[1] = bp[1];
}

// Unified KAN GEMM: A row = [silu(x) (K=IN) | bspline basis (K=8*IN)],
// C = A @ B, B = [Wbase ; Wspline] in fragment-swizzled bf16.
// Each wave owns MT=2 16-row M-tiles (32 rows) to halve B refetch traffic.
// All mt loops are compile-time (MT) so acc[][] stays in VGPRs; the 16-row
// tail is handled by clamping staged row indices and guarding only the
// epilogue with a wave-uniform bound test.
template <int IN, int OUT, bool CONV>
__global__ __launch_bounds__(128) void kan_gemm_kernel(
    const float* __restrict__ X, const int* __restrict__ ei,
    const float* __restrict__ dis, const unsigned short* __restrict__ Bmat,
    const float* __restrict__ knots, float* __restrict__ Y, int Mrows) {
  constexpr int NT = OUT / 16;
  constexpr int MT = 2;                       // 16-row tiles per wave
  __shared__ __align__(16) unsigned short As[4][MT * 16 * 128];
  __shared__ int   colS[4][MT * 16];
  __shared__ int   rowS[4][MT * 16];
  __shared__ float normS[4][MT * 16];

  const int wv   = threadIdx.x >> 5;
  const int lane = threadIdx.x & 31;
  const int tileBase = (blockIdx.x * 4 + wv) * (MT * 16);
  if (tileBase >= Mrows) return;              // per-wave exit, no block barriers
  const int lastRow = Mrows - 1;

  const float t0    = knots[0];
  const float inv_h = 1.f / (knots[1] - knots[0]);

  unsigned short* asw = &As[wv][0];

  if constexpr (CONV) {
    for (int idx = lane; idx < MT * 16; idx += 32) {
      int e = tileBase + idx; e = e > lastRow ? lastRow : e;   // clamp tail
      int r, c; load_edge(ei, e, r, c);
      colS[wv][idx] = c; rowS[wv][idx] = r;
      normS[wv][idx] = dis[r] * dis[c];
    }
    wave_lds_sync();
  }

  v8f zero = {0.f, 0.f, 0.f, 0.f, 0.f, 0.f, 0.f, 0.f};
  v8f acc[MT][NT];
#pragma unroll
  for (int mt = 0; mt < MT; ++mt)
#pragma unroll
    for (int nt = 0; nt < NT; ++nt) acc[mt][nt] = zero;

  const int hi = lane >> 4;
  const int mA = lane & 15;

  // ---- Phase 1: base path, K = IN over silu(x) ----
  for (int idx = lane; idx < MT * 16 * IN; idx += 32) {
    int m = idx / IN, i = idx - m * IN;
    float x;
    if constexpr (CONV) {
      int node = (i < IN / 2) ? colS[wv][m] : rowS[wv][m];
      x = X[(size_t)node * (IN / 2) + (i & (IN / 2 - 1))];
    } else {
      int rowI = tileBase + m; rowI = rowI > lastRow ? lastRow : rowI;
      x = X[(size_t)rowI * IN + i];
    }
    asw[m * 128 + i] = f2bf(x / (1.f + __expf(-x)));
  }
  wave_lds_sync();
#pragma unroll
  for (int kk = 0; kk < IN; kk += 32) {
    Frag16 a[MT];
#pragma unroll
    for (int mt = 0; mt < MT; ++mt)
      load_afrag(a[mt], asw + mt * 16 * 128, mA, hi, kk);
    int kb = kk >> 5;
#pragma unroll
    for (int nt = 0; nt < NT; ++nt) {
      Frag16 b; load_bfrag(b, Bmat, kb, NT, nt, lane);
#pragma unroll
      for (int mt = 0; mt < MT; ++mt)
        acc[mt][nt] = __builtin_amdgcn_wmma_f32_16x16x32_bf16(
            false, a[mt].v, false, b.v, (short)0, acc[mt][nt], false, false);
    }
  }
  wave_lds_sync();

  // ---- Phase 2: spline path, K = 8*IN, chunked 16 features (K=128) ----
  for (int fc = 0; fc < IN; fc += 16) {
    for (int idx = lane; idx < MT * 16 * 16; idx += 32) {
      int m = idx >> 4, ii = idx & 15;
      int i = fc + ii;
      float x;
      if constexpr (CONV) {
        int node = (i < IN / 2) ? colS[wv][m] : rowS[wv][m];
        x = X[(size_t)node * (IN / 2) + (i & (IN / 2 - 1))];
      } else {
        int rowI = tileBase + m; rowI = rowI > lastRow ? lastRow : rowI;
        x = X[(size_t)rowI * IN + i];
      }
      x = fminf(fmaxf(x, -0.99f), 0.99f);
      basis_store(asw + m * 128 + ii * 8, x, t0, inv_h);
    }
    wave_lds_sync();
    const int kg = IN + fc * 8;
#pragma unroll
    for (int kk = 0; kk < 128; kk += 32) {
      Frag16 a[MT];
#pragma unroll
      for (int mt = 0; mt < MT; ++mt)
        load_afrag(a[mt], asw + mt * 16 * 128, mA, hi, kk);
      int kb = (kg + kk) >> 5;
#pragma unroll
      for (int nt = 0; nt < NT; ++nt) {
        Frag16 b; load_bfrag(b, Bmat, kb, NT, nt, lane);
#pragma unroll
        for (int mt = 0; mt < MT; ++mt)
          acc[mt][nt] = __builtin_amdgcn_wmma_f32_16x16x32_bf16(
              false, a[mt].v, false, b.v, (short)0, acc[mt][nt], false, false);
      }
    }
    wave_lds_sync();
  }

  // ---- Epilogue: C/D layout lane->N, vgpr r -> M = r + 8*hi ----
  const int nL   = lane & 15;
  const int mOff = hi * 8;
#pragma unroll
  for (int mt = 0; mt < MT; ++mt) {
    if (tileBase + mt * 16 >= Mrows) continue;   // wave-uniform tail guard
    if constexpr (CONV) {
#pragma unroll
      for (int nt = 0; nt < NT; ++nt)
#pragma unroll
        for (int r = 0; r < 8; ++r) {
          int m = mt * 16 + r + mOff;
          float v = acc[mt][nt][r] * normS[wv][m];
          atomic_add_f32(&Y[(size_t)colS[wv][m] * OUT + nt * 16 + nL], v);
        }
    } else {
#pragma unroll
      for (int nt = 0; nt < NT; ++nt)
#pragma unroll
        for (int r = 0; r < 8; ++r)
          Y[(size_t)(tileBase + mt * 16 + r + mOff) * OUT + nt * 16 + nL] =
              acc[mt][nt][r];
    }
  }
}

// Convert [Wbase(OUTxIN) ; Wspline(OUTxINx8)] f32 -> bf16 in WMMA-B
// fragment-swizzled order: dst[((k/32 * NT + n/16)*32 + lane)*16 + k%16],
// lane = (k%32 >= 16)*16 + n%16.
__global__ void prep_weights_kernel(const float* __restrict__ base,
                                    const float* __restrict__ spline,
                                    unsigned short* __restrict__ dst, int IN, int OUT) {
  int KT = IN * 9;
  int total = KT * OUT;
  int NT = OUT >> 4;
  for (int idx = blockIdx.x * blockDim.x + threadIdx.x; idx < total;
       idx += gridDim.x * blockDim.x) {
    int k = idx / OUT, o = idx - k * OUT;
    float v = (k < IN) ? base[o * IN + k] : spline[(size_t)o * IN * 8 + (k - IN)];
    int kb = k >> 5, koff = k & 31;
    int lane = ((koff >> 4) << 4) | (o & 15);
    int e  = koff & 15;
    int nt = o >> 4;
    dst[(((size_t)(kb * NT + nt) * 32 + lane) << 4) + e] = f2bf(v);
  }
}

__global__ void zero_kernel(float* __restrict__ p, int n) {
  for (int i = blockIdx.x * blockDim.x + threadIdx.x; i < n; i += gridDim.x * blockDim.x)
    p[i] = 0.f;
}

__global__ void degree_kernel(const int* __restrict__ ei, float* __restrict__ deg) {
  int e = blockIdx.x * blockDim.x + threadIdx.x;
  if (e < ETOT) {
    int r, c; load_edge(ei, e, r, c);
    atomic_add_f32(&deg[c], 1.f);
  }
}

__global__ void rsqrt_kernel(const float* __restrict__ deg, float* __restrict__ dis, int n) {
  int i = blockIdx.x * blockDim.x + threadIdx.x;
  if (i < n) {
    float d = deg[i];
    dis[i] = (d > 0.f) ? rsqrtf(d) : 0.f;
  }
}

__global__ void layernorm_kernel(const float* __restrict__ tin, const float* __restrict__ res,
                                 const float* __restrict__ g, const float* __restrict__ b,
                                 float* __restrict__ out, int n, int useRes) {
  int i = blockIdx.x * blockDim.x + threadIdx.x;
  if (i >= n) return;
  float v[64];
  float mu = 0.f;
#pragma unroll
  for (int f = 0; f < 64; ++f) { v[f] = tin[(size_t)i * 64 + f]; mu += v[f]; }
  mu *= (1.f / 64.f);
  float var = 0.f;
#pragma unroll
  for (int f = 0; f < 64; ++f) { float d = v[f] - mu; var += d * d; }
  var *= (1.f / 64.f);
  float inv = rsqrtf(var + 1e-5f);
#pragma unroll
  for (int f = 0; f < 64; ++f) {
    float o = (v[f] - mu) * inv * g[f] + b[f];
    if (useRes) o += res[(size_t)i * 64 + f];
    out[(size_t)i * 64 + f] = o;
  }
}

extern "C" void kernel_launch(void* const* d_in, const int* in_sizes, int n_in,
                              void* d_out, int out_size, void* d_ws, size_t ws_size,
                              hipStream_t stream) {
  (void)in_sizes; (void)n_in; (void)out_size; (void)ws_size;
  const float* x            = (const float*)d_in[0];
  const int*   ei           = (const int*)d_in[1];
  const float* grid         = (const float*)d_in[2];
  const float* in_base_w    = (const float*)d_in[3];
  const float* in_spline_w  = (const float*)d_in[4];
  const float* conv_base_w  = (const float*)d_in[5];
  const float* conv_spline_w= (const float*)d_in[6];
  const float* ln_g         = (const float*)d_in[7];
  const float* ln_b         = (const float*)d_in[8];
  const float* out_base_w   = (const float*)d_in[9];
  const float* out_spline_w = (const float*)d_in[10];
  float* out = (float*)d_out;

  char* ws = (char*)d_ws;
  size_t off = 0;
  auto take = [&](size_t bytes) -> char* {
    char* p = ws + off;
    off = (off + bytes + 255) & ~(size_t)255;
    return p;
  };
  float* hA   = (float*)take((size_t)NN * 64 * 4);
  float* hB   = (float*)take((size_t)NN * 64 * 4);
  float* tmp  = (float*)take((size_t)NN * 64 * 4);
  float* deg  = (float*)take((size_t)NN * 4);
  float* dis  = (float*)take((size_t)NN * 4);
  unsigned short* Bin = (unsigned short*)take((size_t)576 * 64 * 2);
  unsigned short* Bc[3];
  for (int l = 0; l < 3; ++l) Bc[l] = (unsigned short*)take((size_t)1152 * 64 * 2);
  unsigned short* Bout = (unsigned short*)take((size_t)576 * 16 * 2);

  // Weight conversion into fragment-swizzled bf16 (tiny; L2-resident after).
  prep_weights_kernel<<<(576 * 64 + 255) / 256, 256, 0, stream>>>(in_base_w, in_spline_w, Bin, 64, 64);
  for (int l = 0; l < 3; ++l)
    prep_weights_kernel<<<(1152 * 64 + 255) / 256, 256, 0, stream>>>(
        conv_base_w + (size_t)l * 64 * 128, conv_spline_w + (size_t)l * 64 * 128 * 8, Bc[l], 128, 64);
  prep_weights_kernel<<<(576 * 16 + 255) / 256, 256, 0, stream>>>(out_base_w, out_spline_w, Bout, 64, 16);

  // Degree / symmetric norm (same edge list every layer).
  zero_kernel<<<(NN + 255) / 256, 256, 0, stream>>>(deg, NN);
  degree_kernel<<<(ETOT + 255) / 256, 256, 0, stream>>>(ei, deg);
  rsqrt_kernel<<<(NN + 255) / 256, 256, 0, stream>>>(deg, dis, NN);

  const int denseBlocks = (NN + 127) / 128;    // 4 waves x 32 rows per block
  const int convBlocks  = (ETOT + 127) / 128;

  // Input KAN: x[10000,64] -> hA[10000,64]
  kan_gemm_kernel<64, 64, false><<<denseBlocks, 128, 0, stream>>>(
      x, nullptr, nullptr, Bin, grid, hA, NN);

  float* cur = hA;
  float* nxt = hB;
  for (int l = 0; l < 3; ++l) {
    zero_kernel<<<(NN * 64 + 255) / 256, 256, 0, stream>>>(tmp, NN * 64);
    kan_gemm_kernel<128, 64, true><<<convBlocks, 128, 0, stream>>>(
        cur, ei, dis, Bc[l], grid, tmp, ETOT);
    layernorm_kernel<<<(NN + 63) / 64, 64, 0, stream>>>(
        tmp, cur, ln_g + l * 64, ln_b + l * 64, nxt, NN, (l > 0) ? 1 : 0);
    float* t2 = cur; cur = nxt; nxt = t2;
  }

  // Output KAN: cur[10000,64] -> out[10000,16]
  kan_gemm_kernel<64, 16, false><<<denseBlocks, 128, 0, stream>>>(
      cur, nullptr, nullptr, Bout, grid, out, NN);
}
